// MultiHeadAttention_4294967296470
// MI455X (gfx1250) — matrix operand order repaired and checked by reading, not compiled, not run on hardware
//
#include <hip/hip_runtime.h>

typedef __attribute__((ext_vector_type(16))) __bf16 v16bf;
typedef __attribute__((ext_vector_type(8)))  __bf16 v8bf;
typedef __attribute__((ext_vector_type(8)))  float  v8f;
typedef __attribute__((ext_vector_type(4)))  int    v4i;
typedef unsigned short u16;
typedef unsigned int   u32;

#define EMBED 1024
#define NH    16
#define HD    64
#define SEQ   2048

// ---------- optional CDNA5 async global->LDS path (probe via __has_builtin) --
#if __has_builtin(__builtin_amdgcn_global_load_async_to_lds_b128) && \
    __has_builtin(__builtin_amdgcn_s_wait_asynccnt)
#define HAVE_ASYNC_LDS 1
typedef __attribute__((address_space(1))) v4i* gptr_v4i;  // global int4*
typedef __attribute__((address_space(3))) v4i* lptr_v4i;  // LDS int4*
__device__ __forceinline__ void async_g2l_b128(const void* gsrc, void* ldst) {
  gptr_v4i g = (gptr_v4i)(unsigned long long)(uintptr_t)gsrc;
  lptr_v4i l = (lptr_v4i)(u32)(uintptr_t)ldst;  // low 32 bits = LDS offset
  __builtin_amdgcn_global_load_async_to_lds_b128(g, l, 0, 0);
}
#endif

// ---------- bf16 helpers (round-to-nearest-even) ----------
__device__ __forceinline__ u16 f2bf_u16(float f) {
  u32 u = __builtin_bit_cast(u32, f);
  u32 r = u + 0x7FFFu + ((u >> 16) & 1u);
  return (u16)(r >> 16);
}
__device__ __forceinline__ __bf16 f2bf(float f) {
  u16 h = f2bf_u16(f);
  return __builtin_bit_cast(__bf16, h);
}

// ---------- WMMA fragment loaders (CDNA5 ISA §7.12.2 layouts) ----------
// A-frag: 16x32 (MxK) bf16 tile at (row0,k0) of a row-major matrix, ld elems.
// lanes 0-15 hold K = {0..7, 16..23}; lanes 16-31 hold K = {8..15, 24..31}.
__device__ __forceinline__ v16bf load_frag_a(const __bf16* p, int ld, int row0,
                                             int k0, int lane) {
  int l = lane & 15, hi = lane >> 4;
  const __bf16* q = p + (size_t)(row0 + l) * ld + k0 + hi * 8;
  v8bf c0 = *(const v8bf*)q;
  v8bf c1 = *(const v8bf*)(q + 16);
  v16bf a;
#pragma unroll
  for (int i = 0; i < 8; ++i) { a[i] = c0[i]; a[i + 8] = c1[i]; }
  return a;
}

// B-frag: 32x16 (KxN) where B[k][n] = src[n0+n][k0+k], src row-major ld elems
// (source holds the transpose K-major -> two 16B contiguous runs per lane).
// lanes 0-15 (col n) hold K = {0..7, 16..23}; lanes 16-31 hold K = {8..15, 24..31}.
__device__ __forceinline__ v16bf load_frag_b(const __bf16* p, int ld, int n0,
                                             int k0, int lane) {
  int n = lane & 15, hi = lane >> 4;
  const __bf16* q = p + (size_t)(n0 + n) * ld + k0 + hi * 8;
  v8bf c0 = *(const v8bf*)q;
  v8bf c1 = *(const v8bf*)(q + 16);
  v16bf b;
#pragma unroll
  for (int i = 0; i < 8; ++i) { b[i] = c0[i]; b[i + 8] = c1[i]; }
  return b;
}

__device__ __forceinline__ v8f wmma_bf16(v16bf a, v16bf b, v8f c) {
  return __builtin_amdgcn_wmma_f32_16x16x32_bf16(
      /*neg_a=*/false, a, /*neg_b=*/false, b,
      /*c_mod=*/(short)0, c, /*reuse_a=*/false, /*reuse_b=*/false);
}

// ---------- fp32 -> bf16 conversion ----------
__global__ void convert_f32_to_bf16(const float* __restrict__ in,
                                    __bf16* __restrict__ out, int n4) {
  int i = blockIdx.x * blockDim.x + threadIdx.x;
  if (i >= n4) return;
  float4 f = ((const float4*)in)[i];
  u32 lo = (u32)f2bf_u16(f.x) | ((u32)f2bf_u16(f.y) << 16);
  u32 hi = (u32)f2bf_u16(f.z) | ((u32)f2bf_u16(f.w) << 16);
  uint2 o; o.x = lo; o.y = hi;
  ((uint2*)out)[i] = o;
}

// ---------- GEMM: C[m,n] = sum_k A[m,k] * W[n,k] + bias[n] ----------
// Each wave computes a 16x64 strip (4 accumulators, A-frag reused 4x).
// Block: 8 waves -> 16x512 tile.  Grid: (M/16, N/512).
// MODE 0: bf16 out row-major [M,N]
// MODE 1: bf16 out per-head transposed Vt[B,H,Hd,S] (V projection)
// MODE 2: f32  out row-major [M,N] (final output projection)
template <int MODE>
__global__ __launch_bounds__(256) void gemm_bf16_wmma(
    const __bf16* __restrict__ A, const __bf16* __restrict__ W,
    const float* __restrict__ bias, void* __restrict__ Cout, int M, int N,
    int K, int S) {
  __shared__ __bf16 As[16 * EMBED];  // 32 KB A-panel shared by 8 waves
  const int m0 = blockIdx.x * 16;
  const int tid = threadIdx.x;
  const int wave = tid >> 5, lane = tid & 31;

  // stage the 16xK A panel into LDS (async DMA if available)
  {
    const char* src = (const char*)(A + (size_t)m0 * K);
    char* dst = (char*)As;
    const int nchunks = 16 * K * 2 / 16;  // 16-byte chunks
#if defined(HAVE_ASYNC_LDS)
    for (int i = tid; i < nchunks; i += 256)
      async_g2l_b128(src + (size_t)i * 16, dst + (size_t)i * 16);
    __builtin_amdgcn_s_wait_asynccnt(0);
#else
    for (int i = tid; i < nchunks; i += 256)
      ((uint4*)dst)[i] = ((const uint4*)src)[i];
#endif
  }
  __syncthreads();

  const int n0 = blockIdx.y * 512 + wave * 64;
  v8f acc[4] = {{}, {}, {}, {}};
#pragma unroll 2
  for (int k0 = 0; k0 < K; k0 += 32) {
    v16bf a = load_frag_a(As, K, 0, k0, lane);
#pragma unroll
    for (int t = 0; t < 4; ++t) {
      v16bf b = load_frag_b(W, K, n0 + t * 16, k0, lane);
      acc[t] = wmma_bf16(a, b, acc[t]);
    }
  }

  // C element (M = r + 8*hi, N = lane&15) per tile
  const int nl = lane & 15, hi = lane >> 4;
#pragma unroll
  for (int t = 0; t < 4; ++t) {
    const int n = n0 + t * 16 + nl;
    const float bcol = bias[n];
#pragma unroll
    for (int r = 0; r < 8; ++r) {
      int m = m0 + r + 8 * hi;
      float val = acc[t][r] + bcol;
      if (MODE == 0) {
        ((__bf16*)Cout)[(size_t)m * N + n] = f2bf(val);
      } else if (MODE == 1) {
        int bb = m / S, s = m - bb * S;
        int h = n >> 6, d = n & 63;
        ((__bf16*)Cout)[((size_t)(bb * NH + h) * HD + d) * S + s] = f2bf(val);
      } else {
        ((float*)Cout)[(size_t)m * N + n] = val;
      }
    }
  }
}

// ---------- flash attention: one 16-row q-block per wave ----------
// Q,K: bf16 [B,S,D] (head slice h*HD), Vt: bf16 [B,H,Hd,S], O: bf16 [B,S,D]
__global__ __launch_bounds__(256) void attention_wmma(
    const __bf16* __restrict__ Q, const __bf16* __restrict__ Km,
    const __bf16* __restrict__ Vt, __bf16* __restrict__ O) {
  __shared__ __bf16 Pst[8][16 * 32];  // per-wave P staging (C-layout -> A-layout)
  const int tid = threadIdx.x, wave = tid >> 5, lane = tid & 31;
  const int h = blockIdx.y, b = blockIdx.z;
  const int qb = blockIdx.x * 8 + wave;
  const int D = NH * HD;

  const __bf16* qbase = Q + ((size_t)b * SEQ + qb * 16) * D + h * HD;
  const __bf16* kbase = Km + (size_t)b * SEQ * D + h * HD;
  const __bf16* vbase = Vt + (size_t)(b * NH + h) * HD * SEQ;

  // Q A-fragments for d = 0..31 and 32..63, reused across the whole K loop
  const v16bf aq0 = load_frag_a(qbase, D, 0, 0, lane);
  const v16bf aq1 = load_frag_a(qbase, D, 0, 32, lane);

  v8f accO[4] = {{}, {}, {}, {}};
  float mrow[8], lrow[8];  // stats for row (r + 8*(lane>=16))
#pragma unroll
  for (int r = 0; r < 8; ++r) { mrow[r] = -3.0e38f; lrow[r] = 0.0f; }

  __bf16* pst = &Pst[wave][0];

  for (int k0 = 0; k0 < SEQ; k0 += 32) {
    // scores for columns [k0,k0+16) and [k0+16,k0+32): 4 WMMAs
    v8f s0 = {}, s1 = {};
    {
      v16bf b0 = load_frag_b(kbase, D, k0, 0, lane);
      s0 = wmma_bf16(aq0, b0, s0);
      v16bf b1 = load_frag_b(kbase, D, k0, 32, lane);
      s0 = wmma_bf16(aq1, b1, s0);
      v16bf b2 = load_frag_b(kbase, D, k0 + 16, 0, lane);
      s1 = wmma_bf16(aq0, b2, s1);
      v16bf b3 = load_frag_b(kbase, D, k0 + 16, 32, lane);
      s1 = wmma_bf16(aq1, b3, s1);
    }
    // online softmax (reference multiplies scores by sqrt(Hd) = 8)
#pragma unroll
    for (int r = 0; r < 8; ++r) {
      float x0 = s0[r] * 8.0f, x1 = s1[r] * 8.0f;
      float tmax = fmaxf(x0, x1);
#pragma unroll
      for (int sh = 8; sh >= 1; sh >>= 1)
        tmax = fmaxf(tmax, __shfl_xor(tmax, sh, 32));  // stays in 16-lane half
      float mnew = fmaxf(mrow[r], tmax);
      float corr = __expf(mrow[r] - mnew);
      float p0 = __expf(x0 - mnew), p1 = __expf(x1 - mnew);
      float psum = p0 + p1;
#pragma unroll
      for (int sh = 8; sh >= 1; sh >>= 1) psum += __shfl_xor(psum, sh, 32);
      lrow[r] = lrow[r] * corr + psum;
      mrow[r] = mnew;
#pragma unroll
      for (int nt = 0; nt < 4; ++nt) accO[nt][r] *= corr;
      s0[r] = p0; s1[r] = p1;
    }
    // stage P (16 x 32) into this wave's LDS slab, row-major
    {
      const int nl = lane & 15, hi = lane >> 4;
#pragma unroll
      for (int r = 0; r < 8; ++r) {
        int row = r + 8 * hi;
        pst[row * 32 + nl]      = f2bf(s0[r]);
        pst[row * 32 + 16 + nl] = f2bf(s1[r]);
      }
    }
    v16bf ap = load_frag_a(pst, 32, 0, 0, lane);
    // P @ V over the 4 head-dim tiles: 4 WMMAs (Vt is K-major -> contiguous)
#pragma unroll
    for (int nt = 0; nt < 4; ++nt) {
      v16bf bv = load_frag_b(vbase, SEQ, nt * 16, k0, lane);
      accO[nt] = wmma_bf16(ap, bv, accO[nt]);
    }
  }

  // normalize and store bf16 attention output [B,S,D]
  const int nl = lane & 15, hi = lane >> 4;
  __bf16* obase = O + ((size_t)b * SEQ + qb * 16) * D + h * HD;
#pragma unroll
  for (int r = 0; r < 8; ++r) {
    float inv = 1.0f / lrow[r];
    int row = r + 8 * hi;
#pragma unroll
    for (int nt = 0; nt < 4; ++nt)
      obase[(size_t)row * D + nt * 16 + nl] = f2bf(accO[nt][r] * inv);
  }
}

// ---------------------------------------------------------------------------
extern "C" void kernel_launch(void* const* d_in, const int* in_sizes, int n_in,
                              void* d_out, int out_size, void* d_ws,
                              size_t ws_size, hipStream_t stream) {
  (void)in_sizes; (void)n_in; (void)out_size; (void)ws_size;
  const float* query = (const float*)d_in[0];
  const float* key   = (const float*)d_in[1];
  const float* value = (const float*)d_in[2];
  const float* Wq = (const float*)d_in[3];
  const float* bq = (const float*)d_in[4];
  const float* Wk = (const float*)d_in[5];
  const float* bk = (const float*)d_in[6];
  const float* Wv = (const float*)d_in[7];
  const float* bv = (const float*)d_in[8];
  const float* Wo = (const float*)d_in[9];
  const float* bo = (const float*)d_in[10];

  const int B = 2, S = SEQ, D = EMBED, M = B * S;
  const size_t actE = (size_t)M * D;   // 4M elems
  const size_t wE   = (size_t)D * D;   // 1M elems

  char* ws = (char*)d_ws;
  size_t off = 0;
  auto alloc = [&](size_t bytes) {
    void* p = ws + off;
    off += (bytes + 255) & ~(size_t)255;
    return p;
  };
  __bf16* xq = (__bf16*)alloc(actE * 2);
  __bf16* xk = (__bf16*)alloc(actE * 2);
  __bf16* xv = (__bf16*)alloc(actE * 2);
  __bf16* wq = (__bf16*)alloc(wE * 2);
  __bf16* wk = (__bf16*)alloc(wE * 2);
  __bf16* wv = (__bf16*)alloc(wE * 2);
  __bf16* wo = (__bf16*)alloc(wE * 2);
  __bf16* qb = (__bf16*)alloc(actE * 2);
  __bf16* kb = (__bf16*)alloc(actE * 2);
  __bf16* vt = (__bf16*)alloc(actE * 2);
  __bf16* at = (__bf16*)alloc(actE * 2);

  auto cvt = [&](const float* src, __bf16* dst, size_t n) {
    int n4 = (int)(n / 4);
    convert_f32_to_bf16<<<(n4 + 255) / 256, 256, 0, stream>>>(src, dst, n4);
  };
  cvt(query, xq, actE);
  cvt(key,   xk, actE);
  cvt(value, xv, actE);
  cvt(Wq, wq, wE);
  cvt(Wk, wk, wE);
  cvt(Wv, wv, wE);
  cvt(Wo, wo, wE);

  dim3 gemmGrid(M / 16, D / 512);
  gemm_bf16_wmma<0><<<gemmGrid, 256, 0, stream>>>(xq, wq, bq, qb, M, D, D, S);
  gemm_bf16_wmma<0><<<gemmGrid, 256, 0, stream>>>(xk, wk, bk, kb, M, D, D, S);
  gemm_bf16_wmma<1><<<gemmGrid, 256, 0, stream>>>(xv, wv, bv, vt, M, D, D, S);

  dim3 attnGrid(S / 16 / 8, NH, B);
  attention_wmma<<<attnGrid, 256, 0, stream>>>(qb, kb, vt, at);

  gemm_bf16_wmma<2><<<gemmGrid, 256, 0, stream>>>(at, wo, bo, d_out, M, D, D, S);
}
